// Decoder_23922967839349
// MI455X (gfx1250) — matrix operand order repaired and checked by reading, not compiled, not run on hardware
//
#include <hip/hip_runtime.h>
#include <hip/hip_bf16.h>
#include <cstddef>

// ---------------------------------------------------------------------------
// Problem constants (from reference): B=256, T=256, M=512, P=512
// ---------------------------------------------------------------------------
#define BB 256
#define TT 256
#define MM 512
#define PP 512

typedef __attribute__((ext_vector_type(16))) __bf16 v16bf;
typedef __attribute__((ext_vector_type(8)))  float  v8f;

__device__ __forceinline__ unsigned short f32_to_bf16_rne(float x) {
    unsigned int u = __float_as_uint(x);
    unsigned int r = u + 0x7FFFu + ((u >> 16) & 1u);   // round-to-nearest-even
    return (unsigned short)(r >> 16);
}

__device__ __forceinline__ float sigmoidf_(float x) {
    return 1.0f / (1.0f + __expf(-x));
}

// ---------------------------------------------------------------------------
// Fragment loaders per CDNA5 WMMA VGPR layouts (CDNA5 ISA §7.12.2).
// A-matrix 16x32 bf16: lane L -> row = L%16; elements 0..7 hold K = kb..kb+7,
// elements 8..15 hold K = kb+16..kb+23, with kb = kc + (L>=16 ? 8 : 0).
// ---------------------------------------------------------------------------
__device__ __forceinline__ v16bf load_a_frag_f32(const float* __restrict__ A, int lda,
                                                 int row, int kc, int lane) {
    int kb = kc + ((lane & 16) ? 8 : 0);
    const float4* p0 = (const float4*)(A + (size_t)row * lda + kb);
    float4 x0 = p0[0];
    float4 x1 = p0[1];
    const float4* p1 = (const float4*)(A + (size_t)row * lda + kb + 16);
    float4 x2 = p1[0];
    float4 x3 = p1[1];
    union { v16bf v; unsigned short s[16]; } f;
    f.s[0]  = f32_to_bf16_rne(x0.x); f.s[1]  = f32_to_bf16_rne(x0.y);
    f.s[2]  = f32_to_bf16_rne(x0.z); f.s[3]  = f32_to_bf16_rne(x0.w);
    f.s[4]  = f32_to_bf16_rne(x1.x); f.s[5]  = f32_to_bf16_rne(x1.y);
    f.s[6]  = f32_to_bf16_rne(x1.z); f.s[7]  = f32_to_bf16_rne(x1.w);
    f.s[8]  = f32_to_bf16_rne(x2.x); f.s[9]  = f32_to_bf16_rne(x2.y);
    f.s[10] = f32_to_bf16_rne(x2.z); f.s[11] = f32_to_bf16_rne(x2.w);
    f.s[12] = f32_to_bf16_rne(x3.x); f.s[13] = f32_to_bf16_rne(x3.y);
    f.s[14] = f32_to_bf16_rne(x3.z); f.s[15] = f32_to_bf16_rne(x3.w);
    return f.v;
}

// Same A layout, but source already bf16: two contiguous 16B loads, zero VALU.
__device__ __forceinline__ v16bf load_a_frag_bf16(const unsigned short* __restrict__ A, int lda,
                                                  int row, int kc, int lane) {
    int kb = kc + ((lane & 16) ? 8 : 0);
    union { v16bf v; uint4 u[2]; } f;
    f.u[0] = *(const uint4*)(A + (size_t)row * lda + kb);
    f.u[1] = *(const uint4*)(A + (size_t)row * lda + kb + 16);
    return f.v;
}

// B-matrix 32x16 bf16 (column n = lane%16, same K layout as A):
// elements 0..7 hold K = kb..kb+7, elements 8..15 hold K = kb+16..kb+23,
// with kb = kc + (L>=16 ? 8 : 0).
// B is stored as weight matrix W[n, k] row-major (y = x W^T), so a lane's
// 16 K-values are contiguous in memory.
__device__ __forceinline__ v16bf load_b_frag_bf16(const unsigned short* __restrict__ Bw, int ldb,
                                                  int col, int kc, int lane) {
    int kb = kc + ((lane & 16) ? 8 : 0);
    union { v16bf v; uint4 u[2]; } f;
    f.u[0] = *(const uint4*)(Bw + (size_t)col * ldb + kb);
    f.u[1] = *(const uint4*)(Bw + (size_t)col * ldb + kb + 16);
    return f.v;
}

// ---------------------------------------------------------------------------
// WMMA GEMM: D[m,n] = sum_k A1[m,k]*B1[n,k] (+ sum_k A2[m,k]*B2[n,k])
//                     (+ bias1[n]) (+ bias2[n])
// One wave computes a 16x64 output strip (4 consecutive 16x16 N-tiles):
// the A fragment is loaded once per K-chunk and reused by 4 independent
// v_wmma_f32_16x16x32_bf16 issues (hides WMMA->WMMA latency, 4x less A traffic).
// AF32: A operands are f32 converted on load (used only for the one-time UH GEMM).
// ---------------------------------------------------------------------------
template<bool AF32>
__global__ void wmma_gemm_kernel(const void* __restrict__ A1v, int lda1,
                                 const unsigned short* __restrict__ B1, int ldb1, int K1,
                                 const void* __restrict__ A2v, int lda2,
                                 const unsigned short* __restrict__ B2, int ldb2, int K2,
                                 const float* __restrict__ bias1,
                                 const float* __restrict__ bias2,
                                 float* __restrict__ D, int N,
                                 int groupsN, int numWaves) {
    int wave = blockIdx.x * (blockDim.x >> 5) + (threadIdx.x >> 5);
    if (wave >= numWaves) return;
    int lane = threadIdx.x & 31;
    int tm = wave / groupsN;
    int g  = wave - tm * groupsN;

    int row     = tm * 16 + (lane & 15);
    int colBase = g * 64 + (lane & 15);

    v8f c0 = {}, c1 = {}, c2 = {}, c3 = {};

    for (int kc = 0; kc < K1; kc += 32) {
        v16bf a;
        if constexpr (AF32) a = load_a_frag_f32((const float*)A1v, lda1, row, kc, lane);
        else                a = load_a_frag_bf16((const unsigned short*)A1v, lda1, row, kc, lane);
        v16bf b0 = load_b_frag_bf16(B1, ldb1, colBase,      kc, lane);
        v16bf b1 = load_b_frag_bf16(B1, ldb1, colBase + 16, kc, lane);
        v16bf b2 = load_b_frag_bf16(B1, ldb1, colBase + 32, kc, lane);
        v16bf b3 = load_b_frag_bf16(B1, ldb1, colBase + 48, kc, lane);
        c0 = __builtin_amdgcn_wmma_f32_16x16x32_bf16(false, a, false, b0, (short)0, c0, false, false);
        c1 = __builtin_amdgcn_wmma_f32_16x16x32_bf16(false, a, false, b1, (short)0, c1, false, false);
        c2 = __builtin_amdgcn_wmma_f32_16x16x32_bf16(false, a, false, b2, (short)0, c2, false, false);
        c3 = __builtin_amdgcn_wmma_f32_16x16x32_bf16(false, a, false, b3, (short)0, c3, false, false);
    }
    if (A2v != nullptr) {
        for (int kc = 0; kc < K2; kc += 32) {
            v16bf a;
            if constexpr (AF32) a = load_a_frag_f32((const float*)A2v, lda2, row, kc, lane);
            else                a = load_a_frag_bf16((const unsigned short*)A2v, lda2, row, kc, lane);
            v16bf b0 = load_b_frag_bf16(B2, ldb2, colBase,      kc, lane);
            v16bf b1 = load_b_frag_bf16(B2, ldb2, colBase + 16, kc, lane);
            v16bf b2 = load_b_frag_bf16(B2, ldb2, colBase + 32, kc, lane);
            v16bf b3 = load_b_frag_bf16(B2, ldb2, colBase + 48, kc, lane);
            c0 = __builtin_amdgcn_wmma_f32_16x16x32_bf16(false, a, false, b0, (short)0, c0, false, false);
            c1 = __builtin_amdgcn_wmma_f32_16x16x32_bf16(false, a, false, b1, (short)0, c1, false, false);
            c2 = __builtin_amdgcn_wmma_f32_16x16x32_bf16(false, a, false, b2, (short)0, c2, false, false);
            c3 = __builtin_amdgcn_wmma_f32_16x16x32_bf16(false, a, false, b3, (short)0, c3, false, false);
        }
    }

    // C/D layout: VGPR r holds row M = tm*16 + r + (lane>=16 ? 8 : 0), col = lane%16
    int rbase = tm * 16 + ((lane & 16) ? 8 : 0);
    v8f acc[4] = {c0, c1, c2, c3};
    for (int j = 0; j < 4; ++j) {
        int col = colBase + j * 16;
        float bias = 0.0f;
        if (bias1 != nullptr) bias += bias1[col];
        if (bias2 != nullptr) bias += bias2[col];
        for (int r = 0; r < 8; ++r) {
            D[(size_t)(rbase + r) * N + col] = acc[j][r] + bias;
        }
    }
}

// ---------------------------------------------------------------------------
// f32 -> bf16(u16) conversion
// ---------------------------------------------------------------------------
__global__ void convert_bf16_kernel(const float* __restrict__ src,
                                    unsigned short* __restrict__ dst, int n) {
    int i = blockIdx.x * blockDim.x + threadIdx.x;
    if (i < n) dst[i] = f32_to_bf16_rne(src[i]);
}

// ---------------------------------------------------------------------------
// State init: q = [d0 | s0] per batch row (f32 state + bf16 mirror for GEMM A)
// ---------------------------------------------------------------------------
__global__ void init_state_kernel(const float* __restrict__ d0,
                                  const float* __restrict__ s0,
                                  float* __restrict__ q,
                                  unsigned short* __restrict__ qb) {
    int i = blockIdx.x * blockDim.x + threadIdx.x;   // B*P
    int b = i >> 9, p = i & 511;
    float d = d0[i], s = s0[i];
    q[(size_t)b * (2 * PP) + p]        = d;
    q[(size_t)b * (2 * PP) + PP + p]   = s;
    qb[(size_t)b * (2 * PP) + p]       = f32_to_bf16_rne(d);
    qb[(size_t)b * (2 * PP) + PP + p]  = f32_to_bf16_rne(s);
}

// ---------------------------------------------------------------------------
// Attention step: e[b,t'] = tanh(wq[b,:] + UH[b,t',:]) . v ; softmax over t';
// beta written straight into attn output slab; c_t[b,:] = sum_t' beta * H[b,t',:]
// One block per batch element; 256 threads; thread tp owns t'=tp.
// ---------------------------------------------------------------------------
__global__ void attn_step_kernel(const float* __restrict__ wq,
                                 const float* __restrict__ UH,
                                 const float* __restrict__ vvec,
                                 const float* __restrict__ H,
                                 float* __restrict__ beta_out,   // attn_base + t*T
                                 float* __restrict__ c_t,
                                 unsigned short* __restrict__ c_tb) {
    __shared__ float sv[MM];
    __shared__ float swq[MM];
    __shared__ float sbeta[TT];
    __shared__ float sr[TT];

    int b = blockIdx.x;
    int tp = threadIdx.x;

    sv[tp]        = vvec[tp];
    sv[tp + 256]  = vvec[tp + 256];
    swq[tp]       = wq[(size_t)b * MM + tp];
    swq[tp + 256] = wq[(size_t)b * MM + tp + 256];
    __syncthreads();

    const float* uh = UH + ((size_t)b * TT + tp) * MM;
    float e = 0.0f;
    for (int m = 0; m < MM; m += 4) {
        float4 u4 = *(const float4*)(uh + m);
        e += tanhf(swq[m]     + u4.x) * sv[m];
        e += tanhf(swq[m + 1] + u4.y) * sv[m + 1];
        e += tanhf(swq[m + 2] + u4.z) * sv[m + 2];
        e += tanhf(swq[m + 3] + u4.w) * sv[m + 3];
    }

    // softmax over the 256 t' values
    sr[tp] = e;
    __syncthreads();
    for (int s = 128; s > 0; s >>= 1) {
        if (tp < s) sr[tp] = fmaxf(sr[tp], sr[tp + s]);
        __syncthreads();
    }
    float mx = sr[0];
    __syncthreads();
    float ex = __expf(e - mx);
    sr[tp] = ex;
    __syncthreads();
    for (int s = 128; s > 0; s >>= 1) {
        if (tp < s) sr[tp] += sr[tp + s];
        __syncthreads();
    }
    float beta = ex / sr[0];
    sbeta[tp] = beta;
    beta_out[(size_t)b * TT * TT + tp] = beta;
    __syncthreads();

    // context c_t[b, m] = sum_t' beta[t'] * H[b, t', m]
    const float* Hb = H + (size_t)b * TT * MM;
    for (int m = tp; m < MM; m += 256) {
        float acc = 0.0f;
        for (int tt = 0; tt < TT; ++tt) {
            acc += sbeta[tt] * Hb[(size_t)tt * MM + m];
        }
        c_t[(size_t)b * MM + m]  = acc;
        c_tb[(size_t)b * MM + m] = f32_to_bf16_rne(acc);
    }
}

// ---------------------------------------------------------------------------
// LSTM pointwise update. gates layout [B, 4P] in order i,f,g,o.
// q rows are [d(512) | s(512)] (+ bf16 mirror qb for next step's GEMMs).
// On last step also emit out = [d | c_t].
// ---------------------------------------------------------------------------
__global__ void lstm_step_kernel(const float* __restrict__ gates,
                                 float* __restrict__ q,
                                 unsigned short* __restrict__ qb,
                                 const float* __restrict__ c_t,
                                 float* __restrict__ out,
                                 int last) {
    int idx = blockIdx.x * blockDim.x + threadIdx.x;   // B*P
    int b = idx >> 9, p = idx & 511;
    const float* g = gates + (size_t)b * (4 * PP);
    float gi = sigmoidf_(g[p]);
    float gf = sigmoidf_(g[PP + p]);
    float gg = tanhf(g[2 * PP + p]);
    float go = sigmoidf_(g[3 * PP + p]);
    float s_old = q[(size_t)b * (2 * PP) + PP + p];
    float s_new = gf * s_old + gi * gg;
    float d_new = go * tanhf(s_new);
    q[(size_t)b * (2 * PP) + p]        = d_new;
    q[(size_t)b * (2 * PP) + PP + p]   = s_new;
    qb[(size_t)b * (2 * PP) + p]       = f32_to_bf16_rne(d_new);
    qb[(size_t)b * (2 * PP) + PP + p]  = f32_to_bf16_rne(s_new);
    if (last) {
        out[(size_t)b * (PP + MM) + p]       = d_new;
        out[(size_t)b * (PP + MM) + PP + p]  = c_t[(size_t)b * MM + p];
    }
}

// ---------------------------------------------------------------------------
// Host-side orchestration
// ---------------------------------------------------------------------------
extern "C" void kernel_launch(void* const* d_in, const int* in_sizes, int n_in,
                              void* d_out, int out_size, void* d_ws, size_t ws_size,
                              hipStream_t stream) {
    const float* H    = (const float*)d_in[0];   // [B,T,M]
    const float* d0   = (const float*)d_in[1];   // [1,B,P]
    const float* s0   = (const float*)d_in[2];   // [1,B,P]
    const float* W_d  = (const float*)d_in[3];   // [M,2P]
    const float* U_d  = (const float*)d_in[4];   // [M,M]
    const float* v_d  = (const float*)d_in[5];   // [1,M]
    const float* W_ih = (const float*)d_in[6];   // [4P,M]
    const float* W_hh = (const float*)d_in[7];   // [4P,P]
    const float* b_ih = (const float*)d_in[8];   // [4P]
    const float* b_hh = (const float*)d_in[9];   // [4P]

    float* out       = (float*)d_out;                       // [B,1,P+M]
    float* attn_base = out + (size_t)BB * (PP + MM);        // [B,T,T]

    // ---- workspace carve-up (~145 MB) ----
    char* ws = (char*)d_ws;
    size_t off = 0;
    auto carve = [&](size_t bytes) {
        void* p = ws + off;
        off += (bytes + 255) & ~(size_t)255;
        return p;
    };
    float*          UH    = (float*)carve((size_t)BB * TT * MM * 4);       // 134 MB
    unsigned short* U_db  = (unsigned short*)carve((size_t)MM * MM * 2);
    unsigned short* W_db  = (unsigned short*)carve((size_t)MM * 2 * PP * 2);
    unsigned short* W_ihb = (unsigned short*)carve((size_t)4 * PP * MM * 2);
    unsigned short* W_hhb = (unsigned short*)carve((size_t)4 * PP * PP * 2);
    float*          q     = (float*)carve((size_t)BB * 2 * PP * 4);
    unsigned short* qb    = (unsigned short*)carve((size_t)BB * 2 * PP * 2);
    float*          wq    = (float*)carve((size_t)BB * MM * 4);
    float*          c_t   = (float*)carve((size_t)BB * MM * 4);
    unsigned short* c_tb  = (unsigned short*)carve((size_t)BB * MM * 2);
    float*          gates = (float*)carve((size_t)BB * 4 * PP * 4);
    (void)ws_size;

    // ---- weight conversion to bf16 ----
    {
        int n;
        n = MM * MM;
        convert_bf16_kernel<<<(n + 255) / 256, 256, 0, stream>>>(U_d, U_db, n);
        n = MM * 2 * PP;
        convert_bf16_kernel<<<(n + 255) / 256, 256, 0, stream>>>(W_d, W_db, n);
        n = 4 * PP * MM;
        convert_bf16_kernel<<<(n + 255) / 256, 256, 0, stream>>>(W_ih, W_ihb, n);
        n = 4 * PP * PP;
        convert_bf16_kernel<<<(n + 255) / 256, 256, 0, stream>>>(W_hh, W_hhb, n);
    }

    // ---- init recurrent state q = [d0 | s0] ----
    init_state_kernel<<<(BB * PP) / 256, 256, 0, stream>>>(d0, s0, q, qb);

    // ---- UH = H @ U_d^T : [B*T, M] x [M, M]  (time-invariant, once) ----
    {
        int groupsN  = (MM / 16) / 4;                   // 8
        int numWaves = (BB * TT / 16) * groupsN;        // 32768
        wmma_gemm_kernel<true><<<numWaves / 4, 128, 0, stream>>>(
            H, MM, U_db, MM, MM,
            nullptr, 0, nullptr, 0, 0,
            nullptr, nullptr,
            UH, MM, groupsN, numWaves);
    }

    // ---- 256 decode steps ----
    for (int t = 0; t < TT; ++t) {
        // wq = q @ W_d^T : [B, 2P] x [2P, M]
        {
            int groupsN  = (MM / 16) / 4;               // 8
            int numWaves = (BB / 16) * groupsN;         // 128
            wmma_gemm_kernel<false><<<numWaves / 4, 128, 0, stream>>>(
                qb, 2 * PP, W_db, 2 * PP, 2 * PP,
                nullptr, 0, nullptr, 0, 0,
                nullptr, nullptr,
                wq, MM, groupsN, numWaves);
        }

        // attention: scores + softmax + context; beta -> attn[:, t, :]
        attn_step_kernel<<<BB, TT, 0, stream>>>(wq, UH, v_d, H,
                                                attn_base + (size_t)t * TT, c_t, c_tb);

        // gates = c_t @ W_ih^T + d @ W_hh^T + b_ih + b_hh : [B, 4P]
        {
            int groupsN  = ((4 * PP) / 16) / 4;         // 32
            int numWaves = (BB / 16) * groupsN;         // 512
            wmma_gemm_kernel<false><<<numWaves / 4, 128, 0, stream>>>(
                c_tb, MM, W_ihb, MM, MM,
                qb, 2 * PP, W_hhb, PP, PP,              // A2 = d part of qb rows
                b_ih, b_hh,
                gates, 4 * PP, groupsN, numWaves);
        }

        // LSTM pointwise update; final step writes [d | c_t] into out
        lstm_step_kernel<<<(BB * PP) / 256, 256, 0, stream>>>(
            gates, q, qb, c_t, out, (t == TT - 1) ? 1 : 0);
    }
}
